// GATClassifier_48524540510669
// MI455X (gfx1250) — matrix operand order repaired and checked by reading, not compiled, not run on hardware
//
#include <hip/hip_runtime.h>
#include <hip/hip_bf16.h>
#include <stdint.h>

typedef __attribute__((ext_vector_type(16))) _Float16 v16h;
typedef __attribute__((ext_vector_type(8)))  float    v8f;

#define NEG_SLOPE 0.2f

// ---------- helpers ----------

// Total-order encoding of float into uint32 so unsigned atomicMax == float max.
__device__ __forceinline__ unsigned ordF32(float f) {
    unsigned u = __float_as_uint(f);
    return (u & 0x80000000u) ? ~u : (u | 0x80000000u);
}
__device__ __forceinline__ float unordF32(unsigned u) {
    return __uint_as_float((u & 0x80000000u) ? (u ^ 0x80000000u) : ~u);
}

__device__ __forceinline__ void edge_nodes(const int* __restrict__ ei, int E, int e,
                                           int& s, int& d) {
    if (e < E) { s = ei[e]; d = ei[E + e]; }
    else       { s = e - E; d = e - E; }   // self-loop
}

// ---------- generic zero fill ----------
__global__ void zero_f32(float* __restrict__ p, long n) {
    long t = (long)blockIdx.x * blockDim.x + threadIdx.x;
    if (t < n) p[t] = 0.0f;
}

// ---------- WMMA GEMM: C[N,M] = A[N,K] @ B[K,M] (fp32 in/out, f16 compute) ----------
// Block = 8 waves. Each wave computes a 16(rows)x64(cols) strip with 4 WMMAs per
// 32-wide K step. B tile (32 x 64) is staged once per block in LDS as f16 in
// [col][k] order so each lane's operand is one 32B-aligned v16h (2x ds_load_b128).
// Requires N % 16 == 0 (50000 = 3125*16), K % 32 == 0, M % 64 == 0.
__global__ __launch_bounds__(256)
void gemm_wmma_f16(const float* __restrict__ A, const float* __restrict__ B,
                   float* __restrict__ C, int N, int K, int M) {
    __shared__ _Float16 ldsB[64 * 32];            // [col][k] : 4 KB

    const int lane     = threadIdx.x & 31;
    const int wave     = threadIdx.x >> 5;
    const int rowTiles = N >> 4;
    const int rtRaw    = blockIdx.y * 8 + wave;
    const bool active  = rtRaw < rowTiles;
    const int rowTile  = active ? rtRaw : rowTiles - 1;   // clamp; keep barriers uniform
    const int colBase  = blockIdx.x * 64;

    const int halfSel  = lane >> 4;               // 0: lanes 0-15, 1: lanes 16-31
    const int l15      = lane & 15;
    const int arow     = rowTile * 16 + l15;      // A row this lane feeds
    const int koffA    = halfSel * 16;            // A: K rows 0..15 / 16..31
    const int koffB    = halfSel * 16;            // B: K rows 0..15 / 16..31

    // B fill assignment: thread -> (k row, 8 consecutive cols)
    const int fk = threadIdx.x >> 3;              // 0..31
    const int fc = (threadIdx.x & 7) * 8;         // 0,8,...,56

    v8f acc[4] = {};

    for (int kk = 0; kk < K; kk += 32) {
        // ---- stage B[kk..kk+32)[colBase..colBase+64) into LDS as f16 ----
        const float* bg = B + (size_t)(kk + fk) * M + colBase + fc;
        float4 q0 = *(const float4*)(bg);
        float4 q1 = *(const float4*)(bg + 4);
        ldsB[(fc + 0) * 32 + fk] = (_Float16)q0.x;
        ldsB[(fc + 1) * 32 + fk] = (_Float16)q0.y;
        ldsB[(fc + 2) * 32 + fk] = (_Float16)q0.z;
        ldsB[(fc + 3) * 32 + fk] = (_Float16)q0.w;
        ldsB[(fc + 4) * 32 + fk] = (_Float16)q1.x;
        ldsB[(fc + 5) * 32 + fk] = (_Float16)q1.y;
        ldsB[(fc + 6) * 32 + fk] = (_Float16)q1.z;
        ldsB[(fc + 7) * 32 + fk] = (_Float16)q1.w;
        __syncthreads();

        // ---- A operand: lane holds row arow, K = kk+koffA+0..15 ----
        const float* ap = A + (size_t)arow * K + kk + koffA;
        float4 p0 = *(const float4*)(ap);
        float4 p1 = *(const float4*)(ap + 4);
        float4 p2 = *(const float4*)(ap + 8);
        float4 p3 = *(const float4*)(ap + 12);
        v16h av;
        av[0]  = (_Float16)p0.x; av[1]  = (_Float16)p0.y;
        av[2]  = (_Float16)p0.z; av[3]  = (_Float16)p0.w;
        av[4]  = (_Float16)p1.x; av[5]  = (_Float16)p1.y;
        av[6]  = (_Float16)p1.z; av[7]  = (_Float16)p1.w;
        av[8]  = (_Float16)p2.x; av[9]  = (_Float16)p2.y;
        av[10] = (_Float16)p2.z; av[11] = (_Float16)p2.w;
        av[12] = (_Float16)p3.x; av[13] = (_Float16)p3.y;
        av[14] = (_Float16)p3.z; av[15] = (_Float16)p3.w;

        // ---- 4 column tiles, B operand straight from LDS (32B aligned v16h) ----
#pragma unroll
        for (int ct = 0; ct < 4; ++ct) {
            const v16h bv = *(const v16h*)(ldsB + (ct * 16 + l15) * 32 + koffB);
            acc[ct] = __builtin_amdgcn_wmma_f32_16x16x32_f16(
                /*neg_a=*/false, av, /*neg_b=*/false, bv,
                /*c_mod=*/(short)0, acc[ct], /*reuse_a=*/false, /*reuse_b=*/false);
        }
        __syncthreads();
    }

    if (active) {
        // C/D layout: VGPR i -> row M=i (lanes 0-15) or M=8+i (lanes 16-31)
        const int orow = rowTile * 16 + halfSel * 8;
        float* base = C + (size_t)orow * M + colBase + l15;
#pragma unroll
        for (int ct = 0; ct < 4; ++ct) {
            float* cp = base + ct * 16;
#pragma unroll
            for (int i = 0; i < 8; ++i) { *cp = acc[ct][i]; cp += M; }
        }
    }
}

// ---------- attention scores: a_s[n,h] = <h[n,h,:], a_src[h,:]> ----------
__global__ void attn_scores(const float* __restrict__ hfeat,
                            const float* __restrict__ aS, const float* __restrict__ aD,
                            float* __restrict__ os, float* __restrict__ od,
                            int N, int H, int C) {
    int t = blockIdx.x * blockDim.x + threadIdx.x;
    if (t >= N * H) return;
    int n = t / H, hh = t % H;
    const float* hp = hfeat + (size_t)n * H * C + (size_t)hh * C;
    const float* sp = aS + (size_t)hh * C;
    const float* dp = aD + (size_t)hh * C;
    float ss = 0.f, dd = 0.f;
    for (int c = 0; c < C; ++c) { float v = hp[c]; ss += v * sp[c]; dd += v * dp[c]; }
    os[t] = ss; od[t] = dd;
}

// ---------- segment max over edges grouped by dst ----------
__global__ void edge_max_k(const int* __restrict__ ei, int E, int ET,
                           const float* __restrict__ as, const float* __restrict__ ad,
                           unsigned* __restrict__ emax, int H) {
    int t = blockIdx.x * blockDim.x + threadIdx.x;
    if (t >= ET * H) return;
    int e = t / H, hh = t % H;
    int s, d; edge_nodes(ei, E, e, s, d);
    float v = as[s * H + hh] + ad[d * H + hh];
    v = v > 0.f ? v : NEG_SLOPE * v;                 // leaky_relu
    atomicMax(&emax[d * H + hh], ordF32(v));
}

// ---------- segment sum of exp(e - emax[dst]) ----------
__global__ void edge_sum_k(const int* __restrict__ ei, int E, int ET,
                           const float* __restrict__ as, const float* __restrict__ ad,
                           const unsigned* __restrict__ emax, float* __restrict__ denom,
                           int H) {
    int t = blockIdx.x * blockDim.x + threadIdx.x;
    if (t >= ET * H) return;
    int e = t / H, hh = t % H;
    int s, d; edge_nodes(ei, E, e, s, d);
    float v = as[s * H + hh] + ad[d * H + hh];
    v = v > 0.f ? v : NEG_SLOPE * v;
    int di = d * H + hh;
    atomicAdd(&denom[di], __expf(v - unordF32(emax[di])));
}

// ---------- weighted aggregation: out[dst] += alpha * h[src] ----------
// One wave32 per (edge, head); lane covers channels c = lane, lane+32, ...
__global__ __launch_bounds__(256)
void aggregate_k(const int* __restrict__ ei, int E, int ET,
                 const float* __restrict__ hfeat,
                 const float* __restrict__ as, const float* __restrict__ ad,
                 const unsigned* __restrict__ emax, const float* __restrict__ denom,
                 float* __restrict__ out, int H, int C) {
    int wid = blockIdx.x * (blockDim.x >> 5) + (threadIdx.x >> 5);
    if (wid >= ET * H) return;
    int lane = threadIdx.x & 31;
    int e = wid / H, hh = wid % H;
    int s, d; edge_nodes(ei, E, e, s, d);
    float v = as[s * H + hh] + ad[d * H + hh];
    v = v > 0.f ? v : NEG_SLOPE * v;
    int di = d * H + hh;
    float ex    = __expf(v - unordF32(emax[di]));
    float alpha = ex / (denom[di] + 1e-16f);
    const float* hp = hfeat + (size_t)s * H * C + (size_t)hh * C;
    float*       op = out   + (size_t)d * H * C + (size_t)hh * C;
    for (int c = lane; c < C; c += 32)
        atomicAdd(&op[c], alpha * hp[c]);
}

// ---------- bias + ELU (in place) ----------
__global__ void bias_elu_k(float* __restrict__ x, const float* __restrict__ b,
                           long n, int M) {
    long t = (long)blockIdx.x * blockDim.x + threadIdx.x;
    if (t >= n) return;
    float v = x[t] + b[t % M];
    x[t] = v > 0.f ? v : (__expf(v) - 1.0f);
}

// ---------- global mean pool (sum + counts) ----------
__global__ void pool_k(const float* __restrict__ h3, const int* __restrict__ batch,
                       float* __restrict__ pooled, float* __restrict__ cnt,
                       int N, int C) {
    int t = blockIdx.x * blockDim.x + threadIdx.x;
    if (t >= N * C) return;
    int n = t / C, c = t % C;
    int g = batch[n];
    atomicAdd(&pooled[g * C + c], h3[t]);
    if (c == 0) atomicAdd(&cnt[g], 1.0f);
}

// ---------- classifier: out[g,j] = (pooled[g]/cnt[g]) @ Wc + bc ----------
__global__ void classifier_k(const float* __restrict__ pooled, const float* __restrict__ cnt,
                             const float* __restrict__ Wc, const float* __restrict__ bc,
                             float* __restrict__ out, int G, int C, int NC) {
    int t = blockIdx.x * blockDim.x + threadIdx.x;
    if (t >= G * NC) return;
    int g = t / NC, j = t % NC;
    float inv = 1.0f / fmaxf(cnt[g], 1.0f);
    float s = 0.f;
    for (int c = 0; c < C; ++c) s += pooled[g * C + c] * inv * Wc[c * NC + j];
    out[t] = s + bc[j];
}

// ---------- host side ----------
extern "C" void kernel_launch(void* const* d_in, const int* in_sizes, int n_in,
                              void* d_out, int out_size, void* d_ws, size_t ws_size,
                              hipStream_t stream) {
    const float* x      = (const float*)d_in[0];
    const int*   ei     = (const int*)  d_in[1];
    const int*   batch  = (const int*)  d_in[2];
    const float* W1     = (const float*)d_in[3];
    const float* a_src1 = (const float*)d_in[4];
    const float* a_dst1 = (const float*)d_in[5];
    const float* b1     = (const float*)d_in[6];
    const float* W2     = (const float*)d_in[7];
    const float* a_src2 = (const float*)d_in[8];
    const float* a_dst2 = (const float*)d_in[9];
    const float* b2     = (const float*)d_in[10];
    const float* W3     = (const float*)d_in[11];
    const float* a_src3 = (const float*)d_in[12];
    const float* a_dst3 = (const float*)d_in[13];
    const float* b3     = (const float*)d_in[14];
    const float* Wc     = (const float*)d_in[15];
    const float* bc     = (const float*)d_in[16];

    const int F_IN = 128, H = 4, Cc = 64, HC = H * Cc, NC = 200, G = 64;
    const int N  = in_sizes[0] / F_IN;
    const int E  = in_sizes[1] / 2;
    const int ET = E + N;                     // edges incl. self-loops

    // workspace carve-out (256B aligned slices)
    char* wp = (char*)d_ws;
    auto carve = [&](size_t bytes) {
        char* r = wp; wp += (bytes + 255) & ~(size_t)255; return r;
    };
    float*    bufH   = (float*)   carve((size_t)N * HC * 4);   // h = in @ W
    float*    bufA   = (float*)   carve((size_t)N * HC * 4);   // aggregated output
    float*    asw    = (float*)   carve((size_t)N * H * 4);
    float*    adw    = (float*)   carve((size_t)N * H * 4);
    unsigned* emax   = (unsigned*)carve((size_t)N * H * 4);
    float*    denom  = (float*)   carve((size_t)N * H * 4);
    float*    pooled = (float*)   carve((size_t)G * Cc * 4);
    float*    cnt    = (float*)   carve((size_t)G * 4);

    auto blocks = [](long n) { return dim3((unsigned)((n + 255) / 256)); };

    auto run_layer = [&](const float* in, int K, const float* W,
                         const float* aS, const float* aD, const float* b,
                         int h, int c) {
        const int m = h * c;
        dim3 ggrid((unsigned)(m / 64), (unsigned)(((N >> 4) + 7) / 8));
        gemm_wmma_f16<<<ggrid, 256, 0, stream>>>(in, W, bufH, N, K, m);
        attn_scores<<<blocks((long)N * h), 256, 0, stream>>>(bufH, aS, aD, asw, adw, N, h, c);
        zero_f32<<<blocks((long)N * h), 256, 0, stream>>>((float*)emax, (long)N * h); // ord(min)=0
        zero_f32<<<blocks((long)N * h), 256, 0, stream>>>(denom, (long)N * h);
        edge_max_k<<<blocks((long)ET * h), 256, 0, stream>>>(ei, E, ET, asw, adw, emax, h);
        edge_sum_k<<<blocks((long)ET * h), 256, 0, stream>>>(ei, E, ET, asw, adw, emax, denom, h);
        zero_f32<<<blocks((long)N * m), 256, 0, stream>>>(bufA, (long)N * m);
        long waves = (long)ET * h;                           // one wave per (edge, head)
        aggregate_k<<<dim3((unsigned)((waves + 7) / 8)), 256, 0, stream>>>(
            ei, E, ET, bufH, asw, adw, emax, denom, bufA, h, c);
        bias_elu_k<<<blocks((long)N * m), 256, 0, stream>>>(bufA, b, (long)N * m, m);
    };

    run_layer(x,    F_IN, W1, a_src1, a_dst1, b1, H, Cc);   // [N,256]
    run_layer(bufA, HC,   W2, a_src2, a_dst2, b2, H, Cc);   // [N,256]
    run_layer(bufA, HC,   W3, a_src3, a_dst3, b3, 1, Cc);   // [N,64] (H=1, mean==identity)

    zero_f32<<<blocks((long)G * Cc), 256, 0, stream>>>(pooled, (long)G * Cc);
    zero_f32<<<blocks((long)G), 256, 0, stream>>>(cnt, (long)G);
    pool_k<<<blocks((long)N * Cc), 256, 0, stream>>>(bufA, batch, pooled, cnt, N, Cc);
    classifier_k<<<blocks((long)G * NC), 256, 0, stream>>>(pooled, cnt, Wc, bc,
                                                           (float*)d_out, G, Cc, NC);
}
